// CartesianDecomposedAttention_876173328878
// MI455X (gfx1250) — matrix operand order repaired and checked by reading, not compiled, not run on hardware
//
#include <hip/hip_runtime.h>
#include <hip/hip_bf16.h>
#include <stdint.h>

#define DEVI __device__ __forceinline__

typedef __attribute__((ext_vector_type(16))) __bf16 v16bf;
typedef __attribute__((ext_vector_type(8)))  float  v8f;

union V16 { v16bf v; uint4 q[2]; uint32_t u[8]; };
static_assert(sizeof(v16bf) == 32, "v16bf must be 32 bytes");

constexpr int B_  = 8;
constexpr int S_  = 512;
constexpr int D_  = 1024;
constexpr int H_  = 16;
constexpr int DH_ = 64;
constexpr int NTOK = B_ * S_;     // 4096 tokens
constexpr int E3   = 3 * D_;      // 3072 qkv rows
constexpr float SCALE_ = 0.125f;  // 1/sqrt(64)
constexpr float LN1E4_OVER_DH = 0.14391156f; // ln(10000)/64

// element counts
constexpr size_t XN  = (size_t)NTOK * D_;          // 4,194,304
constexpr size_t WQN = (size_t)E3 * D_;            // 3,145,728
constexpr size_t WON = (size_t)D_ * D_;            // 1,048,576
constexpr size_t HN  = (size_t)B_ * H_ * S_ * DH_; // 4,194,304
constexpr size_t OUT_HALF = XN;                    // floats per output plane

DEVI uint16_t f2bf(float f) {
  uint32_t u = __float_as_uint(f);
  u += 0x7FFFu + ((u >> 16) & 1u);          // round-to-nearest-even
  return (uint16_t)(u >> 16);
}

// generic pointer (known to be LDS) -> 32-bit LDS byte offset
DEVI unsigned ldsoff(const void* p) {
  return (unsigned)(unsigned long long)
      (__attribute__((address_space(3))) const void*)p;
}

// issue one async 16B global->LDS transfer (GV mode, ASYNCcnt-tracked)
DEVI void async_b128(unsigned lds_byte, const uint16_t* gptr) {
  const uint64_t ga = (uint64_t)(uintptr_t)gptr;
  asm volatile("global_load_async_to_lds_b128 %0, %1, off"
               :: "v"(lds_byte), "v"(ga) : "memory");
}

DEVI void wait_async0() {
  asm volatile("s_wait_asynccnt 0x0" ::: "memory");
}

// A-matrix tile (16 x K=32, 16-bit). Lane l: M = l&15, K chunks at
// kb..kb+7 and kb+16..kb+23 with kb = (l>>4)*8  (ISA 7.12.2 16-bit A layout).
DEVI v16bf load_a(const uint16_t* p, int ld, int row0, int k0, int lane) {
  const uint16_t* r = p + (size_t)(row0 + (lane & 15)) * ld + k0 + ((lane >> 4) << 3);
  V16 t;
  t.q[0] = *(const uint4*)(r);
  t.q[1] = *(const uint4*)(r + 16);
  return t.v;
}

// B-matrix tile (K=32 x 16, 16-bit). Lane l: N = l&15,
// K chunks at kb..kb+7 and kb+16..kb+23 with kb = (l>>4)*8 (matches A layout).
DEVI v16bf load_b(const uint16_t* p, int ld, int col0, int k0, int lane) {
  const uint16_t* r = p + (size_t)(col0 + (lane & 15)) * ld + k0 + ((lane >> 4) << 3);
  V16 t;
  t.q[0] = *(const uint4*)(r);
  t.q[1] = *(const uint4*)(r + 16);
  return t.v;
}

DEVI v8f wmma_bf(v16bf a, v16bf b, v8f c) {
  return __builtin_amdgcn_wmma_f32_16x16x32_bf16(false, a, false, b, (short)0, c,
                                                 false, false);
}

// ------------------------------------------------ convert + Karatsuba sum
__global__ void cvt3_kernel(const float* __restrict__ sr, const float* __restrict__ si,
                            uint16_t* __restrict__ dr, uint16_t* __restrict__ di,
                            uint16_t* __restrict__ ds_, int n) {
  int i = blockIdx.x * blockDim.x + threadIdx.x;
  if (i < n) {
    const float r = sr[i], im = si[i];
    dr[i]  = f2bf(r);
    di[i]  = f2bf(im);
    ds_[i] = f2bf(r + im);
  }
}

// --------------------------------------------------- QKV projection + RoPE
// 3-WMMA Karatsuba complex GEMM; epilogue applies RoPE and writes planes:
//   q: (qr, qi, qr+qi)  k: (kr, -ki, kr-ki)  v: (vr, vi, vr+vi) transposed.
__global__ __launch_bounds__(128)
void qkv_rope_kernel(const uint16_t* __restrict__ xr, const uint16_t* __restrict__ xi,
                     const uint16_t* __restrict__ xs,
                     const uint16_t* __restrict__ wr, const uint16_t* __restrict__ wi,
                     const uint16_t* __restrict__ ws,
                     uint16_t* __restrict__ q_r, uint16_t* __restrict__ q_i,
                     uint16_t* __restrict__ q_s,
                     uint16_t* __restrict__ k_r, uint16_t* __restrict__ k_n,
                     uint16_t* __restrict__ k_s,
                     uint16_t* __restrict__ vt_r, uint16_t* __restrict__ vt_i,
                     uint16_t* __restrict__ vt_s) {
  const int lane = threadIdx.x & 31, wave = threadIdx.x >> 5;
  const int row0 = blockIdx.x * 32;               // token tile (32 rows)
  const int col0 = (blockIdx.y * 4 + wave) * 32;  // e tile (32 cols)
  const v8f vzero = {};
  v8f P1[2][2], P2[2][2], P3[2][2];
#pragma unroll
  for (int ms = 0; ms < 2; ++ms)
#pragma unroll
    for (int ns = 0; ns < 2; ++ns) { P1[ms][ns] = vzero; P2[ms][ns] = vzero; P3[ms][ns] = vzero; }

#pragma unroll 2
  for (int k0 = 0; k0 < D_; k0 += 32) {
    v16bf ar[2], ai[2], as_[2], br[2], bi[2], bs[2];
#pragma unroll
    for (int ms = 0; ms < 2; ++ms) {
      ar[ms]  = load_a(xr, D_, row0 + 16 * ms, k0, lane);
      ai[ms]  = load_a(xi, D_, row0 + 16 * ms, k0, lane);
      as_[ms] = load_a(xs, D_, row0 + 16 * ms, k0, lane);
    }
#pragma unroll
    for (int ns = 0; ns < 2; ++ns) {
      br[ns] = load_b(wr, D_, col0 + 16 * ns, k0, lane);
      bi[ns] = load_b(wi, D_, col0 + 16 * ns, k0, lane);
      bs[ns] = load_b(ws, D_, col0 + 16 * ns, k0, lane);
    }
    if (k0 + 32 < D_) {
      __builtin_prefetch(wr + (size_t)(col0 + (lane & 15)) * D_ + k0 + 32, 0, 3);
      __builtin_prefetch(ws + (size_t)(col0 + (lane & 15)) * D_ + k0 + 32, 0, 3);
    }
#pragma unroll
    for (int ms = 0; ms < 2; ++ms)
#pragma unroll
      for (int ns = 0; ns < 2; ++ns) {
        P1[ms][ns] = wmma_bf(ar[ms],  br[ns], P1[ms][ns]);
        P2[ms][ns] = wmma_bf(ai[ms],  bi[ns], P2[ms][ns]);
        P3[ms][ns] = wmma_bf(as_[ms], bs[ns], P3[ms][ns]);
      }
  }

#pragma unroll
  for (int ms = 0; ms < 2; ++ms) {
#pragma unroll
    for (int ns = 0; ns < 2; ++ns) {
#pragma unroll
      for (int g = 0; g < 8; ++g) {
        const float p1 = P1[ms][ns][g], p2 = P2[ms][ns][g], p3 = P3[ms][ns][g];
        const float re = p1 - p2;
        const float im = p3 - p1 - p2;
        const int t = row0 + 16 * ms + g + ((lane >> 4) << 3);
        const int e = col0 + 16 * ns + (lane & 15);
        const int b = t >> 9, s = t & (S_ - 1);
        const int dh = e & 63;
        if (e < 2 * D_) {   // q or k: complex rotor e^{i*s*inv_freq[dh]}
          const int h = (e & (D_ - 1)) >> 6;
          float sn, cs;
          __sincosf((float)s * __expf(-(float)dh * LN1E4_OVER_DH), &sn, &cs);
          const float rr = re * cs - im * sn;
          const float ii = re * sn + im * cs;
          const size_t off = (((size_t)b * H_ + h) * S_ + s) * DH_ + dh;
          if (e < D_) {
            q_r[off] = f2bf(rr); q_i[off] = f2bf(ii); q_s[off] = f2bf(rr + ii);
          } else {
            k_r[off] = f2bf(rr); k_n[off] = f2bf(-ii); k_s[off] = f2bf(rr - ii);
          }
        } else {            // v: transposed [b][h][dh][s]
          const int h = (e - 2 * D_) >> 6;
          const size_t off = (((size_t)b * H_ + h) * DH_ + dh) * S_ + s;
          vt_r[off] = f2bf(re); vt_i[off] = f2bf(im); vt_s[off] = f2bf(re + im);
        }
      }
    }
  }
}

// -------------------------------------------------------- flash attention
// One wave = 16 query rows of one (b,h); 8 waves per block share (b,h).
// K/V 32-key blocks are staged into LDS with double-buffered
// GLOBAL_LOAD_ASYNC_TO_LDS_B128 (ASYNCcnt), overlapping copy with compute.
__global__ __launch_bounds__(256)
void attn_kernel(const uint16_t* __restrict__ q_r, const uint16_t* __restrict__ q_i,
                 const uint16_t* __restrict__ q_s,
                 const uint16_t* __restrict__ k_r, const uint16_t* __restrict__ k_n,
                 const uint16_t* __restrict__ k_s,
                 const uint16_t* __restrict__ vt_r, const uint16_t* __restrict__ vt_i,
                 const uint16_t* __restrict__ vt_s,
                 uint16_t* __restrict__ o_r, uint16_t* __restrict__ o_i,
                 uint16_t* __restrict__ o_s) {
  __shared__ uint16_t kbuf[2][3][32][64];   // (kr,-ki,kr-ki) x 32 keys x 64 dh
  __shared__ uint16_t vbuf[2][3][64][32];   // (vr,vi,vr+vi) x 64 dh x 32 keys
  __shared__ uint16_t lwr[8][16][32];
  __shared__ uint16_t lwi[8][16][32];
  __shared__ uint16_t lws[8][16][32];

  const int tid = threadIdx.x;
  const int lane = tid & 31, wave = tid >> 5;
  const int h = blockIdx.y, b = blockIdx.z;
  const int row0 = blockIdx.x * 128 + wave * 16;        // query rows in [0,S)
  const size_t hoff = ((size_t)b * H_ + h) * (size_t)S_ * DH_;
  const uint16_t* qrh = q_r + hoff;  const uint16_t* qih = q_i + hoff;
  const uint16_t* qsh = q_s + hoff;
  const uint16_t* kp0 = k_r + hoff;  const uint16_t* kp1 = k_n + hoff;
  const uint16_t* kp2 = k_s + hoff;
  const uint16_t* vp0 = vt_r + hoff; const uint16_t* vp1 = vt_i + hoff;
  const uint16_t* vp2 = vt_s + hoff;

  const unsigned kbase = ldsoff(&kbuf[0][0][0][0]);   // 12288 B per buffer
  const unsigned vbase = ldsoff(&vbuf[0][0][0][0]);

  // cooperative async copy of one 32-key K/V block into LDS buffer `buf`
  auto issue_copy = [&](int kb, int buf) {
#pragma unroll
    for (int t = 0; t < 3; ++t) {           // K planes: contiguous 4KB each
      const int idx = tid + 256 * t;        // 0..767
      const int plane = idx >> 8, rem = idx & 255;
      const uint16_t* gp = (plane == 0) ? kp0 : (plane == 1) ? kp1 : kp2;
      async_b128(kbase + buf * 12288 + plane * 4096 + rem * 16,
                 gp + (size_t)kb * DH_ + rem * 8);
    }
#pragma unroll
    for (int t = 0; t < 3; ++t) {           // V planes: 64 rows of 64B
      const int idx = tid + 256 * t;
      const int plane = idx >> 8, rem = idx & 255;
      const int dh = rem >> 2, ch = rem & 3;
      const uint16_t* gp = (plane == 0) ? vp0 : (plane == 1) ? vp1 : vp2;
      async_b128(vbase + buf * 12288 + plane * 4096 + dh * 64 + ch * 16,
                 gp + (size_t)dh * S_ + kb + ch * 8);
    }
  };

  v16bf aqr[2], aqi[2], aqs[2];
#pragma unroll
  for (int c = 0; c < 2; ++c) {
    aqr[c] = load_a(qrh, DH_, row0, 32 * c, lane);
    aqi[c] = load_a(qih, DH_, row0, 32 * c, lane);
    aqs[c] = load_a(qsh, DH_, row0, 32 * c, lane);
  }

  const v8f vzero = {};
  v8f Pr1[4], Pr2[4], Pr3[4];               // PV Karatsuba accumulators
#pragma unroll
  for (int nt = 0; nt < 4; ++nt) { Pr1[nt] = vzero; Pr2[nt] = vzero; Pr3[nt] = vzero; }
  float mrow[8], lsum[8];
#pragma unroll
  for (int g = 0; g < 8; ++g) { mrow[g] = -1e30f; lsum[g] = 0.0f; }

  // prologue: stage block 0
  issue_copy(0, 0);
  wait_async0();
  __syncthreads();

  for (int kb = 0; kb < S_; kb += 32) {          // 32 keys per step
    const int buf = (kb >> 5) & 1;
    if (kb + 32 < S_) issue_copy(kb + 32, buf ^ 1);   // overlap next copy

    const uint16_t* kk0 = &kbuf[buf][0][0][0];
    const uint16_t* kk1 = &kbuf[buf][1][0][0];
    const uint16_t* kk2 = &kbuf[buf][2][0][0];
    const uint16_t* vv0 = &vbuf[buf][0][0][0];
    const uint16_t* vv1 = &vbuf[buf][1][0][0];
    const uint16_t* vv2 = &vbuf[buf][2][0][0];

    v8f S1[2], S2[2], S3[2];
#pragma unroll
    for (int sub = 0; sub < 2; ++sub) { S1[sub] = vzero; S2[sub] = vzero; S3[sub] = vzero; }
#pragma unroll
    for (int sub = 0; sub < 2; ++sub) {
#pragma unroll
      for (int c = 0; c < 2; ++c) {
        v16bf bkr = load_b(kk0, DH_, sub * 16, 32 * c, lane);
        v16bf bkn = load_b(kk1, DH_, sub * 16, 32 * c, lane);
        v16bf bks = load_b(kk2, DH_, sub * 16, 32 * c, lane);
        S1[sub] = wmma_bf(aqr[c], bkr, S1[sub]);
        S2[sub] = wmma_bf(aqi[c], bkn, S2[sub]);
        S3[sub] = wmma_bf(aqs[c], bks, S3[sub]);
      }
    }
    v8f Sr[2], Si[2];
#pragma unroll
    for (int sub = 0; sub < 2; ++sub)
#pragma unroll
      for (int g = 0; g < 8; ++g) {
        const float p1 = S1[sub][g], p2 = S2[sub][g], p3 = S3[sub][g];
        Sr[sub][g] = (p1 - p2) * SCALE_;
        Si[sub][g] = (p3 - p1 - p2) * SCALE_;
      }

    // online softmax stats (row = g + 8*(lane>>4); cols across 16-lane half)
    float alpha[8];
#pragma unroll
    for (int g = 0; g < 8; ++g) {
      float v = fmaxf(Sr[0][g], Sr[1][g]);
      v = fmaxf(v, __shfl_xor(v, 1));
      v = fmaxf(v, __shfl_xor(v, 2));
      v = fmaxf(v, __shfl_xor(v, 4));
      v = fmaxf(v, __shfl_xor(v, 8));
      const float mn = fmaxf(mrow[g], v);
      alpha[g] = __expf(mrow[g] - mn);
      mrow[g] = mn;
    }
#pragma unroll
    for (int nt = 0; nt < 4; ++nt)
#pragma unroll
      for (int g = 0; g < 8; ++g) {
        Pr1[nt][g] *= alpha[g]; Pr2[nt][g] *= alpha[g]; Pr3[nt][g] *= alpha[g];
      }

    float psum[8];
#pragma unroll
    for (int g = 0; g < 8; ++g) psum[g] = 0.0f;
#pragma unroll
    for (int sub = 0; sub < 2; ++sub) {
#pragma unroll
      for (int g = 0; g < 8; ++g) {
        const float p = __expf(Sr[sub][g] - mrow[g]);
        float sn, cs;
        __sincosf(Si[sub][g], &sn, &cs);
        psum[g] += p;
        const int mr = g + ((lane >> 4) << 3);
        const int nc = (lane & 15) + 16 * sub;
        const float wr_ = p * cs, wi_ = p * sn;
        lwr[wave][mr][nc] = f2bf(wr_);
        lwi[wave][mr][nc] = f2bf(wi_);
        lws[wave][mr][nc] = f2bf(wr_ + wi_);
      }
    }
#pragma unroll
    for (int g = 0; g < 8; ++g) {
      float v = psum[g];
      v += __shfl_xor(v, 1);
      v += __shfl_xor(v, 2);
      v += __shfl_xor(v, 4);
      v += __shfl_xor(v, 8);
      lsum[g] = lsum[g] * alpha[g] + v;
    }
    __syncthreads();                     // C-layout -> A-layout via LDS
    V16 awr, awi, aws;
    const int ma = lane & 15, kba = (lane >> 4) << 3;
#pragma unroll
    for (int g = 0; g < 8; ++g) {
      const int k = kba + ((g & 3) << 1) + ((g >> 2) << 4);
      awr.u[g] = *(const uint32_t*)&lwr[wave][ma][k];
      awi.u[g] = *(const uint32_t*)&lwi[wave][ma][k];
      aws.u[g] = *(const uint32_t*)&lws[wave][ma][k];
    }

#pragma unroll
    for (int nt = 0; nt < 4; ++nt) {     // PV: V tiles from LDS [dh][key]
      v16bf bvr = load_b(vv0, 32, nt * 16, 0, lane);
      v16bf bvi = load_b(vv1, 32, nt * 16, 0, lane);
      v16bf bvs = load_b(vv2, 32, nt * 16, 0, lane);
      Pr1[nt] = wmma_bf(awr.v, bvr, Pr1[nt]);
      Pr2[nt] = wmma_bf(awi.v, bvi, Pr2[nt]);
      Pr3[nt] = wmma_bf(aws.v, bvs, Pr3[nt]);
    }

    wait_async0();                       // next block staged
    __syncthreads();                     // all reads of current buffers done
  }

  float inv[8];
#pragma unroll
  for (int g = 0; g < 8; ++g) inv[g] = 1.0f / lsum[g];
#pragma unroll
  for (int nt = 0; nt < 4; ++nt) {
#pragma unroll
    for (int g = 0; g < 8; ++g) {
      const float p1 = Pr1[nt][g], p2 = Pr2[nt][g], p3 = Pr3[nt][g];
      const float vr = (p1 - p2) * inv[g];
      const float vi = (p3 - p1 - p2) * inv[g];
      const int s = row0 + g + ((lane >> 4) << 3);
      const int d = h * DH_ + nt * 16 + (lane & 15);
      const size_t off = ((size_t)b * S_ + s) * D_ + d;
      o_r[off] = f2bf(vr);
      o_i[off] = f2bf(vi);
      o_s[off] = f2bf(vr + vi);     // Karatsuba sum plane for output GEMM
    }
  }
}

// ------------------------------------------------------- output projection
__global__ __launch_bounds__(128)
void oproj_kernel(const uint16_t* __restrict__ ar_, const uint16_t* __restrict__ ai_,
                  const uint16_t* __restrict__ as_,
                  const uint16_t* __restrict__ wr, const uint16_t* __restrict__ wi,
                  const uint16_t* __restrict__ ws,
                  float* __restrict__ out) {
  const int lane = threadIdx.x & 31, wave = threadIdx.x >> 5;
  const int row0 = blockIdx.x * 32;
  const int col0 = (blockIdx.y * 4 + wave) * 32;
  const v8f vzero = {};
  v8f P1[2][2], P2[2][2], P3[2][2];
#pragma unroll
  for (int ms = 0; ms < 2; ++ms)
#pragma unroll
    for (int ns = 0; ns < 2; ++ns) { P1[ms][ns] = vzero; P2[ms][ns] = vzero; P3[ms][ns] = vzero; }

#pragma unroll 2
  for (int k0 = 0; k0 < D_; k0 += 32) {
    v16bf a1[2], a2[2], a3[2], b1[2], b2[2], b3[2];
#pragma unroll
    for (int ms = 0; ms < 2; ++ms) {
      a1[ms] = load_a(ar_, D_, row0 + 16 * ms, k0, lane);
      a2[ms] = load_a(ai_, D_, row0 + 16 * ms, k0, lane);
      a3[ms] = load_a(as_, D_, row0 + 16 * ms, k0, lane);
    }
#pragma unroll
    for (int ns = 0; ns < 2; ++ns) {
      b1[ns] = load_b(wr, D_, col0 + 16 * ns, k0, lane);
      b2[ns] = load_b(wi, D_, col0 + 16 * ns, k0, lane);
      b3[ns] = load_b(ws, D_, col0 + 16 * ns, k0, lane);
    }
    if (k0 + 32 < D_) {
      __builtin_prefetch(wr + (size_t)(col0 + (lane & 15)) * D_ + k0 + 32, 0, 3);
      __builtin_prefetch(ws + (size_t)(col0 + (lane & 15)) * D_ + k0 + 32, 0, 3);
    }
#pragma unroll
    for (int ms = 0; ms < 2; ++ms)
#pragma unroll
      for (int ns = 0; ns < 2; ++ns) {
        P1[ms][ns] = wmma_bf(a1[ms], b1[ns], P1[ms][ns]);
        P2[ms][ns] = wmma_bf(a2[ms], b2[ns], P2[ms][ns]);
        P3[ms][ns] = wmma_bf(a3[ms], b3[ns], P3[ms][ns]);
      }
  }
#pragma unroll
  for (int ms = 0; ms < 2; ++ms)
#pragma unroll
    for (int ns = 0; ns < 2; ++ns)
#pragma unroll
      for (int g = 0; g < 8; ++g) {
        const float p1 = P1[ms][ns][g], p2 = P2[ms][ns][g], p3 = P3[ms][ns][g];
        const int t = row0 + 16 * ms + g + ((lane >> 4) << 3);
        const int e = col0 + 16 * ns + (lane & 15);
        out[(size_t)t * D_ + e]            = p1 - p2;
        out[OUT_HALF + (size_t)t * D_ + e] = p3 - p1 - p2;
      }
}

// ------------------------------------------------------------------ launch
extern "C" void kernel_launch(void* const* d_in, const int* in_sizes, int n_in,
                              void* d_out, int out_size, void* d_ws, size_t ws_size,
                              hipStream_t stream) {
  const float* xr     = (const float*)d_in[0];
  const float* xi     = (const float*)d_in[1];
  const float* wqkv_r = (const float*)d_in[2];
  const float* wqkv_i = (const float*)d_in[3];
  const float* wo_r   = (const float*)d_in[4];
  const float* wo_i   = (const float*)d_in[5];
  float* out = (float*)d_out;

  uint16_t* p = (uint16_t*)d_ws;
  uint16_t* xb_r  = p;              uint16_t* xb_i  = xb_r + XN;   uint16_t* xb_s  = xb_i + XN;
  uint16_t* wqb_r = xb_s + XN;      uint16_t* wqb_i = wqb_r + WQN; uint16_t* wqb_s = wqb_i + WQN;
  uint16_t* wob_r = wqb_s + WQN;    uint16_t* wob_i = wob_r + WON; uint16_t* wob_s = wob_i + WON;
  uint16_t* q_r  = wob_s + WON;     uint16_t* q_i  = q_r + HN;     uint16_t* q_s  = q_i + HN;
  uint16_t* k_r  = q_s + HN;        uint16_t* k_n  = k_r + HN;     uint16_t* k_s  = k_n + HN;
  uint16_t* vt_r = k_s + HN;        uint16_t* vt_i = vt_r + HN;    uint16_t* vt_s = vt_i + HN;
  uint16_t* o_r  = vt_s + HN;       uint16_t* o_i  = o_r + HN;     uint16_t* o_s  = o_i + HN;

  // 1) f32 -> bf16 planes (r, i, r+i)
  cvt3_kernel<<<(int)((XN + 255) / 256), 256, 0, stream>>>(xr, xi, xb_r, xb_i, xb_s, (int)XN);
  cvt3_kernel<<<(int)((WQN + 255) / 256), 256, 0, stream>>>(wqkv_r, wqkv_i, wqb_r, wqb_i, wqb_s, (int)WQN);
  cvt3_kernel<<<(int)((WON + 255) / 256), 256, 0, stream>>>(wo_r, wo_i, wob_r, wob_i, wob_s, (int)WON);

  // 2) QKV projection (Karatsuba complex GEMM) + RoPE + head reshape
  qkv_rope_kernel<<<dim3(NTOK / 32, E3 / 128), 128, 0, stream>>>(
      xb_r, xb_i, xb_s, wqb_r, wqb_i, wqb_s,
      q_r, q_i, q_s, k_r, k_n, k_s, vt_r, vt_i, vt_s);

  // 3) flash attention with async-LDS staged K/V and complex phase softmax
  attn_kernel<<<dim3(S_ / 128, H_, B_), 256, 0, stream>>>(
      q_r, q_i, q_s, k_r, k_n, k_s, vt_r, vt_i, vt_s, o_r, o_i, o_s);

  // 4) output projection (Karatsuba complex GEMM), f32 results to d_out
  oproj_kernel<<<dim3(NTOK / 32, D_ / 128), 128, 0, stream>>>(
      o_r, o_i, o_s, wob_r, wob_i, wob_s, out);
}
